// PredictedLDDTHead_80092550136257
// MI455X (gfx1250) — matrix operand order repaired and checked by reading, not compiled, not run on hardware
//
#include <hip/hip_runtime.h>
#include <hip/hip_bf16.h>
#include <math.h>

// ---------------------------------------------------------------------------
// Invariant Point Attention + transition + pLDDT head for MI455X (gfx1250).
// wave32, WMMA f32_16x16x32_f16 everywhere a GEMM shape appears.
//
// Input order (setup_inputs dict order, params flattened in insertion order):
//  0 act[768,384] 1 static_feat_2d[768,768,128] 2 sequence_mask[768]
//  3 affine[768,7] 4 aatype (unused)
//  5 wq 6 wk 7 wv 8 wqp 9 wkp 10 wvp 11 wb 12 gamma 13 wo 14 bo
// 15 ln1_g 16 ln1_b 17 wt1 18 bt1 19 wt2 20 bt2 21 wt3 22 bt3
// 23 ln2_g 24 ln2_b 25 lnin_g 26 lnin_b 27 w0 28 b0 29 w1 30 b1 31 wl 32 bl
// ---------------------------------------------------------------------------

#define NN    768
#define CACT  384
#define CPAIR 128
#define HH    12
#define HPAD  16
#define OCAT  2112   // 192 o_s + 288 op_l + 96 op_n + 1536 o_pr

typedef __attribute__((ext_vector_type(16))) _Float16 v16h;
typedef __attribute__((ext_vector_type(8)))  _Float16 v8h;
typedef __attribute__((ext_vector_type(4)))  _Float16 v4h;
typedef __attribute__((ext_vector_type(8)))  float    v8f;
typedef __attribute__((ext_vector_type(4)))  float    v4f;

__device__ static inline v8f vzero8() {
  v8f z;
#pragma unroll
  for (int i = 0; i < 8; ++i) z[i] = 0.f;
  return z;
}

__device__ static inline v8f wmma16(v16h a, v16h b, v8f c) {
  // D = A(16x32 f16) * B(32x16 f16) + C(16x16 f32)
  return __builtin_amdgcn_wmma_f32_16x16x32_f16(false, a, false, b, (short)0, c,
                                                false, false);
}

// A fragment (16x32, 16-bit): lane m=L&15 holds k = 16*(L>>4) + e (contiguous).
// `row` points at element k=0 of row m (contiguous k).
__device__ static inline v16h afrag_f16(const _Float16* row, int k0, int lane) {
  int b = (lane >> 4) << 4;
  v8h lo = *(const v8h*)(row + k0 + b);
  v8h hi = *(const v8h*)(row + k0 + b + 8);
  v16h a;
#pragma unroll
  for (int i = 0; i < 8; ++i) { a[i] = lo[i]; a[8 + i] = hi[i]; }
  return a;
}

__device__ static inline v16h afrag_f32(const float* row, int k0, int lane) {
  int b = (lane >> 4) << 4;
  const v4f* p0 = (const v4f*)(row + k0 + b);
  const v4f* p1 = (const v4f*)(row + k0 + b + 8);
  v4f x0 = p0[0], x1 = p0[1], y0 = p1[0], y1 = p1[1];
  v16h a;
#pragma unroll
  for (int i = 0; i < 4; ++i) {
    a[i]      = (_Float16)x0[i];
    a[4 + i]  = (_Float16)x1[i];
    a[8 + i]  = (_Float16)y0[i];
    a[12 + i] = (_Float16)y1[i];
  }
  return a;
}

// B fragment (32x16, 16-bit): lane n=L&15 holds k = 16*(L>>4) + e (contiguous).
// `row` points at k=0 of B^T row n (i.e. column n of B, contiguous in k).
__device__ static inline v16h bfrag_f16(const _Float16* row, int k0, int lane) {
  int kb = (lane >> 4) << 4;
  v8h lo = *(const v8h*)(row + k0 + kb);
  v8h hi = *(const v8h*)(row + k0 + kb + 8);
  v16h b;
#pragma unroll
  for (int i = 0; i < 8; ++i) { b[i] = lo[i]; b[8 + i] = hi[i]; }
  return b;
}

__device__ static inline v16h bfrag_f32(const float* row, int k0, int lane) {
  int kb = (lane >> 4) << 4;
  const v4f* p = (const v4f*)(row + k0 + kb);
  v16h b;
#pragma unroll
  for (int j = 0; j < 4; ++j) {
    v4f x = p[j];
#pragma unroll
    for (int i = 0; i < 4; ++i) b[j * 4 + i] = (_Float16)x[i];
  }
  return b;
}

// ---------------------------------------------------------------------------
// Projections: q,k,v + point projections moved to global frame, plus the
// f16-packed operands the WMMA attention kernels consume.
// ---------------------------------------------------------------------------
__global__ __launch_bounds__(192) void k_proj(
    const float* __restrict__ act, const float* __restrict__ affine,
    const float* __restrict__ wq, const float* __restrict__ wk,
    const float* __restrict__ wv_, const float* __restrict__ wqp,
    const float* __restrict__ wkp, const float* __restrict__ wvp,
    float* __restrict__ qproj,          // [NN][192]
    _Float16* __restrict__ kscal,       // [NN][192]
    _Float16* __restrict__ kpkn,        // [NN][160]: 144 kp_g | 12 kn | 4 zero
    _Float16* __restrict__ vcatT,       // [HH][48][NN]: 16 v | 24 vp_g | 8 zero
    float* __restrict__ qp_g,           // [NN][144]
    float* __restrict__ Rout)           // [NN][9]
{
  __shared__ float ar[CACT];
  __shared__ float raw[1152];
  __shared__ float Rs[9];
  __shared__ float kpg[144];
  const int q = blockIdx.x, t = threadIdx.x;

  for (int i = t; i < CACT; i += 192) ar[i] = act[(size_t)q * CACT + i];
  if (t == 0) {
    float qw = affine[q * 7 + 0], qx = affine[q * 7 + 1];
    float qy = affine[q * 7 + 2], qz = affine[q * 7 + 3];
    float inv = rsqrtf(qw * qw + qx * qx + qy * qy + qz * qz + 1e-12f);
    qw *= inv; qx *= inv; qy *= inv; qz *= inv;
    Rs[0] = 1.f - 2.f * (qy * qy + qz * qz); Rs[1] = 2.f * (qx * qy - qw * qz);
    Rs[2] = 2.f * (qx * qz + qw * qy);       Rs[3] = 2.f * (qx * qy + qw * qz);
    Rs[4] = 1.f - 2.f * (qx * qx + qz * qz); Rs[5] = 2.f * (qy * qz - qw * qx);
    Rs[6] = 2.f * (qx * qz - qw * qy);       Rs[7] = 2.f * (qy * qz + qw * qx);
    Rs[8] = 1.f - 2.f * (qx * qx + qy * qy);
#pragma unroll
    for (int i = 0; i < 9; ++i) Rout[q * 9 + i] = Rs[i];
  }
  __syncthreads();

#pragma unroll
  for (int j = 0; j < 6; ++j) {
    int col = t + j * 192;
    const float* W; int lc, ldw;
    if      (col < 192) { W = wq;  lc = col;       ldw = 192; }
    else if (col < 384) { W = wk;  lc = col - 192; ldw = 192; }
    else if (col < 576) { W = wv_; lc = col - 384; ldw = 192; }
    else if (col < 720) { W = wqp; lc = col - 576; ldw = 144; }
    else if (col < 864) { W = wkp; lc = col - 720; ldw = 144; }
    else                { W = wvp; lc = col - 864; ldw = 288; }
    float s = 0.f;
    for (int c = 0; c < CACT; ++c) s += ar[c] * W[(size_t)c * ldw + lc];
    raw[col] = s;
  }
  __syncthreads();

  const float tx = affine[q * 7 + 4], ty = affine[q * 7 + 5], tz = affine[q * 7 + 6];

  // scalar q / k / v outputs
  {
    qproj[(size_t)q * 192 + t] = raw[t];
    kscal[(size_t)q * 192 + t] = (_Float16)raw[192 + t];
    int h = t >> 4, c = t & 15;
    vcatT[((size_t)h * 48 + c) * NN + q] = (_Float16)raw[384 + t];
  }
  if (t < 48) {                       // qp -> global frame
    const float* l = raw + 576 + t * 3;
    float g0 = Rs[0] * l[0] + Rs[1] * l[1] + Rs[2] * l[2] + tx;
    float g1 = Rs[3] * l[0] + Rs[4] * l[1] + Rs[5] * l[2] + ty;
    float g2 = Rs[6] * l[0] + Rs[7] * l[1] + Rs[8] * l[2] + tz;
    float* o = qp_g + (size_t)q * 144 + t * 3;
    o[0] = g0; o[1] = g1; o[2] = g2;
  } else if (t < 96) {                // kp -> global frame
    int i2 = t - 48;
    const float* l = raw + 720 + i2 * 3;
    float g0 = Rs[0] * l[0] + Rs[1] * l[1] + Rs[2] * l[2] + tx;
    float g1 = Rs[3] * l[0] + Rs[4] * l[1] + Rs[5] * l[2] + ty;
    float g2 = Rs[6] * l[0] + Rs[7] * l[1] + Rs[8] * l[2] + tz;
    kpg[i2 * 3 + 0] = g0; kpg[i2 * 3 + 1] = g1; kpg[i2 * 3 + 2] = g2;
    _Float16* o = kpkn + (size_t)q * 160 + i2 * 3;
    o[0] = (_Float16)g0; o[1] = (_Float16)g1; o[2] = (_Float16)g2;
  } else {                            // vp -> global frame
    int i3 = t - 96, h = i3 >> 3, p = i3 & 7;
    const float* l = raw + 864 + i3 * 3;
    float g0 = Rs[0] * l[0] + Rs[1] * l[1] + Rs[2] * l[2] + tx;
    float g1 = Rs[3] * l[0] + Rs[4] * l[1] + Rs[5] * l[2] + ty;
    float g2 = Rs[6] * l[0] + Rs[7] * l[1] + Rs[8] * l[2] + tz;
    vcatT[((size_t)h * 48 + 16 + p * 3 + 0) * NN + q] = (_Float16)g0;
    vcatT[((size_t)h * 48 + 16 + p * 3 + 1) * NN + q] = (_Float16)g1;
    vcatT[((size_t)h * 48 + 16 + p * 3 + 2) * NN + q] = (_Float16)g2;
  }
  if (t < 96) {                       // zero vcatT pad columns 40..47
    int h = t >> 3, cc = 40 + (t & 7);
    vcatT[((size_t)h * 48 + cc) * NN + q] = (_Float16)0.f;
  }
  __syncthreads();
  if (t < HH) {                       // kn = sum |kp_g|^2 over points
    float s = 0.f;
#pragma unroll
    for (int i = 0; i < 12; ++i) { float v = kpg[t * 12 + i]; s += v * v; }
    kpkn[(size_t)q * 160 + 144 + t] = (_Float16)s;
  } else if (t < 16) {
    kpkn[(size_t)q * 160 + 144 + t] = (_Float16)0.f;  // pad cols 156..159
  }
}

// ---------------------------------------------------------------------------
// Fused logits (b2d + qk + point term) + softmax.  One block per query row q.
// Single WMMA chain over K = 128(sf·wb) + 192(k·δq) + 160(kp·δqp, kn·δ) = 480.
// The per-(q,h) constant qn term cancels in softmax and is omitted.
// ---------------------------------------------------------------------------
__global__ __launch_bounds__(64) void k_logits(
    const float* __restrict__ sf, const float* __restrict__ wb,
    const float* __restrict__ gamma, const float* __restrict__ qproj,
    const float* __restrict__ qp_g, const float* __restrict__ mask,
    const _Float16* __restrict__ kscal, const _Float16* __restrict__ kpkn,
    _Float16* __restrict__ a_f16)      // [NN][HPAD][NN]
{
  __shared__ __align__(16) _Float16 Bt[16 * 480];
  __shared__ float lg[HH * NN];
  __shared__ __align__(16) _Float16 tile[2][16 * CPAIR];
  const int q = blockIdx.x, t = threadIdx.x;
  const int lane = t & 31, wid = t >> 5;

  const float wpair  = 0.5773502692f;            // sqrt(1/3)
  const float wscal  = 0.1443375673f;            // sqrt(1/48)
  const float wpoint = 0.1360827635f;            // sqrt(2/108)

  for (int i = t; i < 16 * 480; i += 64) Bt[i] = (_Float16)0.f;
  __syncthreads();
  for (int i = t; i < HH * CPAIR; i += 64) {     // seg1: pair bias weights
    int h = i / CPAIR, c = i % CPAIR;
    Bt[h * 480 + c] = (_Float16)(wpair * wb[c * HH + h]);
  }
  for (int i = t; i < 192; i += 64) {            // seg2: block-diag scalar q
    int h = i >> 4;
    Bt[h * 480 + 128 + i] = (_Float16)(wscal * qproj[(size_t)q * 192 + i]);
  }
  for (int i = t; i < 144; i += 64) {            // seg3: block-diag point q
    int h = i / 12;
    float g = log1pf(expf(gamma[h]));
    Bt[h * 480 + 320 + i] = (_Float16)(wpoint * g * qp_g[(size_t)q * 144 + i]);
  }
  if (t < HH) {                                  // seg3: kn diagonal
    float g = log1pf(expf(gamma[t]));
    Bt[t * 480 + 464 + t] = (_Float16)(-0.5f * wpoint * g);
  }
  __syncthreads();

  for (int it = 0; it < 24; ++it) {
    const int kt = it * 2 + wid;                 // k-tile for this wave
    const int k0g = kt * 16;
    _Float16* tl = tile[wid];
    const float* src = sf + ((size_t)q * NN + k0g) * CPAIR;
    // stage 16x128 f32 tile -> f16 LDS with float4 loads (global_load_b128)
    for (int i = lane; i < (16 * CPAIR) / 4; i += 32) {
      int row = i >> 5, c4 = (i & 31) * 4;
      v4f x = *(const v4f*)(src + (size_t)row * CPAIR + c4);
      v4h hv;
#pragma unroll
      for (int j = 0; j < 4; ++j) hv[j] = (_Float16)x[j];
      *(v4h*)(tl + row * CPAIR + c4) = hv;
    }
    // prefetch this wave's next tile (hides HBM latency behind the WMMA chain)
    if (it < 23) {
      const float* nxt = sf + ((size_t)q * NN + k0g + 32) * CPAIR;
      __builtin_prefetch(nxt + (size_t)lane * 64, 0, 1);
      __builtin_prefetch(nxt + 2048 + (size_t)lane * 64, 0, 1);
    }
    __syncthreads();

    v8f acc = vzero8();
    const _Float16* arow_sf = tl + (lane & 15) * CPAIR;
    const _Float16* brow    = Bt + (lane & 15) * 480;
#pragma unroll
    for (int ch = 0; ch < 4; ++ch)               // pair features (K=128)
      acc = wmma16(afrag_f16(arow_sf, ch * 32, lane),
                   bfrag_f16(brow, ch * 32, lane), acc);
    const _Float16* arow_k = kscal + (size_t)(k0g + (lane & 15)) * 192;
#pragma unroll
    for (int ch = 0; ch < 6; ++ch)               // scalar qk (K=192)
      acc = wmma16(afrag_f16(arow_k, ch * 32, lane),
                   bfrag_f16(brow, 128 + ch * 32, lane), acc);
    const _Float16* arow_p = kpkn + (size_t)(k0g + (lane & 15)) * 160;
#pragma unroll
    for (int ch = 0; ch < 5; ++ch)               // point term (K=160)
      acc = wmma16(afrag_f16(arow_p, ch * 32, lane),
                   bfrag_f16(brow, 320 + ch * 32, lane), acc);

    int h = lane & 15;
    if (h < HH) {
#pragma unroll
      for (int r = 0; r < 8; ++r)
        lg[h * NN + k0g + r + ((lane >> 4) << 3)] = acc[r];
    }
    __syncthreads();
  }

  // softmax over k for 12 heads (6 heads per wave)
  const float mq = mask[q];
  for (int h = wid * 6; h < wid * 6 + 6; ++h) {
    float mx = -1e30f;
    for (int k = lane; k < NN; k += 32) {
      float l = lg[h * NN + k] + (mq * mask[k] - 1.f) * 1e9f;
      lg[h * NN + k] = l;
      mx = fmaxf(mx, l);
    }
#pragma unroll
    for (int off = 16; off; off >>= 1) mx = fmaxf(mx, __shfl_xor(mx, off, 32));
    float sm = 0.f;
    for (int k = lane; k < NN; k += 32) {
      float e = __expf(lg[h * NN + k] - mx);
      lg[h * NN + k] = e;
      sm += e;
    }
#pragma unroll
    for (int off = 16; off; off >>= 1) sm += __shfl_xor(sm, off, 32);
    float inv = 1.f / sm;
    for (int k = lane; k < NN; k += 32)
      a_f16[((size_t)q * HPAD + h) * NN + k] = (_Float16)(lg[h * NN + k] * inv);
  }
  for (int i = t; i < 4 * NN; i += 64)           // zero pad heads 12..15
    a_f16[((size_t)q * HPAD + HH) * NN + i] = (_Float16)0.f;
}

// ---------------------------------------------------------------------------
// o_s and op_g:  per (q-tile, head) GEMM  a[q,:,h] @ [v_h | vp_h]  (K=768).
// ---------------------------------------------------------------------------
__global__ __launch_bounds__(32) void k_attn_av(
    const _Float16* __restrict__ a_f16, const _Float16* __restrict__ vcatT,
    float* __restrict__ o_cat, float* __restrict__ opg)
{
  const int qt = blockIdx.x, h = blockIdx.y, lane = threadIdx.x;
  v8f acc0 = vzero8(), acc1 = vzero8(), acc2 = vzero8();
  const _Float16* arow = a_f16 + ((size_t)(qt * 16 + (lane & 15)) * HPAD + h) * NN;
  const _Float16* b0r = vcatT + ((size_t)h * 48 + 0  + (lane & 15)) * NN;
  const _Float16* b1r = vcatT + ((size_t)h * 48 + 16 + (lane & 15)) * NN;
  const _Float16* b2r = vcatT + ((size_t)h * 48 + 32 + (lane & 15)) * NN;
  for (int ch = 0; ch < 24; ++ch) {
    v16h a = afrag_f16(arow, ch * 32, lane);
    acc0 = wmma16(a, bfrag_f16(b0r, ch * 32, lane), acc0);
    acc1 = wmma16(a, bfrag_f16(b1r, ch * 32, lane), acc1);
    acc2 = wmma16(a, bfrag_f16(b2r, ch * 32, lane), acc2);
  }
  const int n = lane & 15;
#pragma unroll
  for (int r = 0; r < 8; ++r) {
    int q = qt * 16 + r + ((lane >> 4) << 3);
    o_cat[(size_t)q * OCAT + h * 16 + n] = acc0[r];
    opg[((size_t)q * HH + h) * 24 + n] = acc1[r];
    if (n < 8) opg[((size_t)q * HH + h) * 24 + 16 + n] = acc2[r];
  }
}

// op_l = R^T (op_g - t), op_n = |op_l|
__global__ __launch_bounds__(96) void k_finalize(
    const float* __restrict__ opg, const float* __restrict__ R,
    const float* __restrict__ affine, float* __restrict__ o_cat)
{
  const int q = blockIdx.x, t = threadIdx.x;
  const int h = t >> 3, pp = t & 7;
  const float* g = opg + ((size_t)q * HH + h) * 24 + pp * 3;
  const float* r = R + q * 9;
  float dx = g[0] - affine[q * 7 + 4];
  float dy = g[1] - affine[q * 7 + 5];
  float dz = g[2] - affine[q * 7 + 6];
  float l0 = r[0] * dx + r[3] * dy + r[6] * dz;
  float l1 = r[1] * dx + r[4] * dy + r[7] * dz;
  float l2 = r[2] * dx + r[5] * dy + r[8] * dz;
  float* oc = o_cat + (size_t)q * OCAT;
  oc[192 + (h * 8 + pp) * 3 + 0] = l0;
  oc[192 + (h * 8 + pp) * 3 + 1] = l1;
  oc[192 + (h * 8 + pp) * 3 + 2] = l2;
  oc[480 + h * 8 + pp] = sqrtf(l0 * l0 + l1 * l1 + l2 * l2 + 1e-8f);
}

// ---------------------------------------------------------------------------
// o_pr: per query q,  D[c,h] = sum_k sf[q,k,c] * a[q,k,h]  via WMMA with the
// pair tile transposed through LDS (second and last streaming pass over sf).
// ---------------------------------------------------------------------------
__global__ __launch_bounds__(128) void k_opr(
    const float* __restrict__ sf, const _Float16* __restrict__ a_f16,
    float* __restrict__ o_cat)
{
  __shared__ __align__(16) _Float16 tl[CPAIR * 40];  // [c][k-chunk of 32], padded
  const int q = blockIdx.x, t = threadIdx.x;
  const int lane = t & 31, wid = t >> 5;
  v8f acc[2] = { vzero8(), vzero8() };
  const _Float16* brow = a_f16 + ((size_t)q * HPAD + (lane & 15)) * NN;

  for (int ch = 0; ch < 24; ++ch) {
    const int k0g = ch * 32;
    __syncthreads();
    const float* src = sf + ((size_t)q * NN + k0g) * CPAIR;
    // stage 32x128 f32 -> f16 LDS transposed, float4 loads (global_load_b128)
    for (int i = t; i < (32 * CPAIR) / 4; i += 128) {
      int kk = i >> 5, c4 = (i & 31) * 4;
      v4f x = *(const v4f*)(src + (size_t)kk * CPAIR + c4);
      tl[(c4 + 0) * 40 + kk] = (_Float16)x[0];
      tl[(c4 + 1) * 40 + kk] = (_Float16)x[1];
      tl[(c4 + 2) * 40 + kk] = (_Float16)x[2];
      tl[(c4 + 3) * 40 + kk] = (_Float16)x[3];
    }
    if (ch < 23) {                    // prefetch next chunk
      const float* nxt = src + 32 * CPAIR;
      __builtin_prefetch(nxt + (size_t)t * 64, 0, 1);
      __builtin_prefetch(nxt + 8192 + (size_t)t * 64, 0, 1);
    }
    __syncthreads();
    v16h b = bfrag_f16(brow, k0g, lane);
#pragma unroll
    for (int m = 0; m < 2; ++m) {
      const _Float16* arow = tl + ((wid * 2 + m) * 16 + (lane & 15)) * 40;
      acc[m] = wmma16(afrag_f16(arow, 0, lane), b, acc[m]);
    }
  }
  const int h = lane & 15;
  if (h < HH) {
#pragma unroll
    for (int m = 0; m < 2; ++m)
#pragma unroll
      for (int r = 0; r < 8; ++r) {
        int c = (wid * 2 + m) * 16 + r + ((lane >> 4) << 3);
        o_cat[(size_t)q * OCAT + 576 + h * CPAIR + c] = acc[m][r];
      }
  }
}

// ---------------------------------------------------------------------------
// Generic tiled WMMA GEMM: C[M,N] = A[M,K] @ Bt[Npad,K]^T (+bias, optional relu)
// A/Bt f32 in memory, converted to f16 per fragment.  K % 32 == 0.
// Two n-tiles per block (grid.y = Npad/32) so each A fragment is used twice.
// ---------------------------------------------------------------------------
__global__ __launch_bounds__(32) void k_gemm(
    const float* __restrict__ A, const float* __restrict__ Bt,
    const float* __restrict__ bias, float* __restrict__ C,
    int N, int K, int lda, int ldbt, int ldc, int relu)
{
  const int mt = blockIdx.x, nt = blockIdx.y * 2, lane = threadIdx.x;
  const float* arow  = A + (size_t)(mt * 16 + (lane & 15)) * lda;
  const float* brow0 = Bt + (size_t)(nt * 16 + (lane & 15)) * ldbt;
  const float* brow1 = brow0 + (size_t)16 * ldbt;
  v8f acc0 = vzero8(), acc1 = vzero8();
  for (int k0 = 0; k0 < K; k0 += 32) {
    v16h a = afrag_f32(arow, k0, lane);
    acc0 = wmma16(a, bfrag_f32(brow0, k0, lane), acc0);
    acc1 = wmma16(a, bfrag_f32(brow1, k0, lane), acc1);
  }
#pragma unroll
  for (int half = 0; half < 2; ++half) {
    const int n = (nt + half) * 16 + (lane & 15);
    v8f acc = half ? acc1 : acc0;
    if (n < N) {
      float bz = bias ? bias[n] : 0.f;
#pragma unroll
      for (int r = 0; r < 8; ++r) {
        int m = mt * 16 + r + ((lane >> 4) << 3);
        float v = acc[r] + bz;
        if (relu) v = fmaxf(v, 0.f);
        C[(size_t)m * ldc + n] = v;
      }
    }
  }
}

__global__ void k_transpose(const float* __restrict__ W, float* __restrict__ Bt,
                            int K, int N, int Npad) {
  int i = blockIdx.x * 256 + threadIdx.x;
  if (i < Npad * K) {
    int n = i / K, k = i % K;
    Bt[i] = (n < N) ? W[(size_t)k * N + n] : 0.f;
  }
}

// out = LayerNorm(x (+ y)) * g + b over last dim 384
__global__ __launch_bounds__(128) void k_add_ln(
    const float* __restrict__ x, const float* __restrict__ y,
    const float* __restrict__ g, const float* __restrict__ b,
    float* __restrict__ out)
{
  __shared__ float red[8];
  const int q = blockIdx.x, t = threadIdx.x;
  float v[3], s = 0.f, s2 = 0.f;
#pragma unroll
  for (int j = 0; j < 3; ++j) {
    int c = t + j * 128;
    float val = x[(size_t)q * CACT + c] + (y ? y[(size_t)q * CACT + c] : 0.f);
    v[j] = val; s += val; s2 += val * val;
  }
#pragma unroll
  for (int off = 16; off; off >>= 1) {
    s += __shfl_xor(s, off, 32);
    s2 += __shfl_xor(s2, off, 32);
  }
  const int lane = t & 31, wid = t >> 5;
  if (lane == 0) { red[wid] = s; red[4 + wid] = s2; }
  __syncthreads();
  s = red[0] + red[1] + red[2] + red[3];
  s2 = red[4] + red[5] + red[6] + red[7];
  float mean = s / (float)CACT;
  float var = s2 / (float)CACT - mean * mean;
  float inv = rsqrtf(var + 1e-5f);
#pragma unroll
  for (int j = 0; j < 3; ++j) {
    int c = t + j * 128;
    out[(size_t)q * CACT + c] = (v[j] - mean) * inv * g[c] + b[c];
  }
}

// ---------------------------------------------------------------------------
extern "C" void kernel_launch(void* const* d_in, const int* in_sizes, int n_in,
                              void* d_out, int out_size, void* d_ws, size_t ws_size,
                              hipStream_t stream) {
  const float* act    = (const float*)d_in[0];
  const float* sf     = (const float*)d_in[1];
  const float* mask   = (const float*)d_in[2];
  const float* affine = (const float*)d_in[3];
  const float* wq  = (const float*)d_in[5];
  const float* wk  = (const float*)d_in[6];
  const float* wv  = (const float*)d_in[7];
  const float* wqp = (const float*)d_in[8];
  const float* wkp = (const float*)d_in[9];
  const float* wvp = (const float*)d_in[10];
  const float* wb  = (const float*)d_in[11];
  const float* gamma = (const float*)d_in[12];
  const float* wo  = (const float*)d_in[13];
  const float* bo  = (const float*)d_in[14];
  const float* ln1g = (const float*)d_in[15];
  const float* ln1b = (const float*)d_in[16];
  const float* wt1 = (const float*)d_in[17];
  const float* bt1 = (const float*)d_in[18];
  const float* wt2 = (const float*)d_in[19];
  const float* bt2 = (const float*)d_in[20];
  const float* wt3 = (const float*)d_in[21];
  const float* bt3 = (const float*)d_in[22];
  const float* ln2g = (const float*)d_in[23];
  const float* ln2b = (const float*)d_in[24];
  const float* lning = (const float*)d_in[25];
  const float* lninb = (const float*)d_in[26];
  const float* w0 = (const float*)d_in[27];
  const float* b0 = (const float*)d_in[28];
  const float* w1 = (const float*)d_in[29];
  const float* b1 = (const float*)d_in[30];
  const float* wl = (const float*)d_in[31];
  const float* bl = (const float*)d_in[32];

  char* ws = (char*)d_ws;
  size_t off = 0;
  auto take = [&](size_t bytes) -> void* {
    void* p = ws + off;
    off = (off + bytes + 255) & ~(size_t)255;
    return p;
  };
  float*     Rbuf  = (float*)take((size_t)NN * 9 * 4);
  float*     qproj = (float*)take((size_t)NN * 192 * 4);
  float*     qp_g  = (float*)take((size_t)NN * 144 * 4);
  _Float16*  kscal = (_Float16*)take((size_t)NN * 192 * 2);
  _Float16*  kpkn  = (_Float16*)take((size_t)NN * 160 * 2);
  _Float16*  vcatT = (_Float16*)take((size_t)HH * 48 * NN * 2);
  _Float16*  a16   = (_Float16*)take((size_t)NN * HPAD * NN * 2);
  float*     opg   = (float*)take((size_t)NN * HH * 24 * 4);
  float*     o_cat = (float*)take((size_t)NN * OCAT * 4);
  float*     woT   = (float*)take((size_t)CACT * OCAT * 4);
  float*     wt1T  = (float*)take((size_t)CACT * CACT * 4);
  float*     wt2T  = (float*)take((size_t)CACT * CACT * 4);
  float*     wt3T  = (float*)take((size_t)CACT * CACT * 4);
  float*     w0T   = (float*)take((size_t)128 * CACT * 4);
  float*     w1T   = (float*)take((size_t)128 * 128 * 4);
  float*     wlT   = (float*)take((size_t)64 * 128 * 4);
  float*     tatt  = (float*)take((size_t)NN * CACT * 4);
  float*     a1    = (float*)take((size_t)NN * CACT * 4);
  float*     t1    = (float*)take((size_t)NN * CACT * 4);
  float*     t2    = (float*)take((size_t)NN * CACT * 4);
  float*     t3    = (float*)take((size_t)NN * CACT * 4);
  float*     a2    = (float*)take((size_t)NN * CACT * 4);
  float*     h0    = (float*)take((size_t)NN * CACT * 4);
  float*     h1    = (float*)take((size_t)NN * 128 * 4);
  float*     h2    = (float*)take((size_t)NN * 128 * 4);

  // weight transposes (Bt layout [Npad][K])
  auto tr = [&](const float* W, float* Bt, int K, int N, int Npad) {
    int tot = Npad * K;
    k_transpose<<<(tot + 255) / 256, 256, 0, stream>>>(W, Bt, K, N, Npad);
  };
  tr(wo,  woT,  OCAT, CACT, CACT);
  tr(wt1, wt1T, CACT, CACT, CACT);
  tr(wt2, wt2T, CACT, CACT, CACT);
  tr(wt3, wt3T, CACT, CACT, CACT);
  tr(w0,  w0T,  CACT, 128, 128);
  tr(w1,  w1T,  128,  128, 128);
  tr(wl,  wlT,  128,  50,  64);

  k_proj<<<NN, 192, 0, stream>>>(act, affine, wq, wk, wv, wqp, wkp, wvp,
                                 qproj, kscal, kpkn, vcatT, qp_g, Rbuf);
  k_logits<<<NN, 64, 0, stream>>>(sf, wb, gamma, qproj, qp_g, mask,
                                  kscal, kpkn, a16);
  k_attn_av<<<dim3(NN / 16, HH), 32, 0, stream>>>(a16, vcatT, o_cat, opg);
  k_finalize<<<NN, 96, 0, stream>>>(opg, Rbuf, affine, o_cat);
  k_opr<<<NN, 128, 0, stream>>>(sf, a16, o_cat);

  // attn_out = o_cat @ wo + bo ; a1 = LN(act + attn_out)
  k_gemm<<<dim3(NN / 16, CACT / 32), 32, 0, stream>>>(
      o_cat, woT, bo, tatt, CACT, OCAT, OCAT, OCAT, CACT, 0);
  k_add_ln<<<NN, 128, 0, stream>>>(act, tatt, ln1g, ln1b, a1);

  // transition MLP + LN
  k_gemm<<<dim3(NN / 16, CACT / 32), 32, 0, stream>>>(
      a1, wt1T, bt1, t1, CACT, CACT, CACT, CACT, CACT, 1);
  k_gemm<<<dim3(NN / 16, CACT / 32), 32, 0, stream>>>(
      t1, wt2T, bt2, t2, CACT, CACT, CACT, CACT, CACT, 1);
  k_gemm<<<dim3(NN / 16, CACT / 32), 32, 0, stream>>>(
      t2, wt3T, bt3, t3, CACT, CACT, CACT, CACT, CACT, 0);
  k_add_ln<<<NN, 128, 0, stream>>>(a1, t3, ln2g, ln2b, a2);

  // pLDDT head
  k_add_ln<<<NN, 128, 0, stream>>>(a2, (const float*)nullptr, lning, lninb, h0);
  k_gemm<<<dim3(NN / 16, 128 / 32), 32, 0, stream>>>(
      h0, w0T, b0, h1, 128, CACT, CACT, CACT, 128, 1);
  k_gemm<<<dim3(NN / 16, 128 / 32), 32, 0, stream>>>(
      h1, w1T, b1, h2, 128, 128, 128, 128, 128, 1);
  k_gemm<<<dim3(NN / 16, 64 / 32), 32, 0, stream>>>(
      h2, wlT, bl, (float*)d_out, 50, 128, 128, 128, 50, 0);
}
